// PINO_img_21517786153388
// MI455X (gfx1250) — matrix operand order repaired and checked by reading, not compiled, not run on hardware
//
#include <hip/hip_runtime.h>
#include <hip/hip_fp16.h>

using h16 = _Float16;
typedef __attribute__((ext_vector_type(16))) _Float16 v16h;
typedef __attribute__((ext_vector_type(8)))  _Float16 v8h;
typedef __attribute__((ext_vector_type(8)))  float    v8f;
typedef __attribute__((ext_vector_type(4))) unsigned int u32x4;
typedef __attribute__((ext_vector_type(8))) int i32x8;
typedef __attribute__((ext_vector_type(4))) int i32x4;

#if __has_builtin(__builtin_amdgcn_tensor_load_to_lds)
#define USE_TDM 1
#else
#define USE_TDM 0
#endif

static constexpr int Bn = 32, Hh = 128, Wd = 256, HID = 64, LYR = 4;
static constexpr int KX = 24, KY = 12;            // retained modes (12 low + 12 high kx)
static constexpr int HW = Hh * Wd;                // 32768
static constexpr long NPIX = (long)Bn * HW;       // 1048576
static constexpr float PI2 = 6.28318530717958647692f;

__device__ __forceinline__ float geluf(float x) {
  return 0.5f * x * (1.0f + erff(x * 0.707106781186547524f));
}

__device__ __forceinline__ v8f wmma_acc(v16h a, v16h b, v8f c) {
  return __builtin_amdgcn_wmma_f32_16x16x32_f16(false, a, false, b, (short)0, c, false, false);
}

__device__ __forceinline__ v16h cat8(v8h lo, v8h hi) {
  return __builtin_shufflevector(lo, hi, 0, 1, 2, 3, 4, 5, 6, 7, 8, 9, 10, 11, 12, 13, 14, 15);
}

// ---- fast-path fragment loads for contiguous row-major data ----
// A fragment 16x32 (MxK): lanes 0-15 (M=lane) hold K 0..15; lanes 16-31 K 16..31.
__device__ __forceinline__ v16h loadA(const h16* A, long lda, int m0, int k0, int lane) {
  const h16* p = A + (long)(m0 + (lane & 15)) * lda + k0 + ((lane < 16) ? 0 : 16);
  v8h lo = *(const v8h*)p;
  v8h hi = *(const v8h*)(p + 8);
  return cat8(lo, hi);
}
// B fragment 32x16 (KxN): lanes 0-15 (N=lane) hold K 0..15; lanes 16-31 K 16..31.
__device__ __forceinline__ v16h loadB(const h16* Bm, long ldb, long n0, int k0, int lane) {
  const h16* p = Bm + (n0 + (lane & 15)) * ldb + k0 + ((lane < 16) ? 0 : 16);
  return *(const v16h*)p;
}

// ---- generic (gather) fragment loads ----
template <class F>
__device__ __forceinline__ v16h load_fragA(int lane, F f) {
  int m = lane & 15, kb = (lane < 16) ? 0 : 8;
  v16h v;
#pragma unroll
  for (int i = 0; i < 8; ++i) v[i] = f(m, kb + i);
#pragma unroll
  for (int i = 0; i < 8; ++i) v[8 + i] = f(m, 16 + kb + i);
  return v;
}
template <class F>
__device__ __forceinline__ v16h load_fragB(int lane, F f) {
  int n = lane & 15, kb = (lane < 16) ? 0 : 16;
  v16h v;
#pragma unroll
  for (int i = 0; i < 16; ++i) v[i] = f(n, kb + i);
  return v;
}
// C/D: VGPR r -> (M = m0 + r or r+8, N = lane&15); scalar store with transform
template <class F>
__device__ __forceinline__ void store_frag(int lane, v8f acc, F f) {
  int n = lane & 15, mb = (lane < 16) ? 0 : 8;
#pragma unroll
  for (int r = 0; r < 8; ++r) f(n, mb + r, acc[r]);
}
// packed f16 store: 8 contiguous m per lane, with per-element transform f(n,m,v)->float
template <class F>
__device__ __forceinline__ void store_vec(h16* base, long ld, long n0, int m0, int lane, v8f acc,
                                          F f) {
  int n = lane & 15, mb = (lane < 16) ? 0 : 8;
  h16* dst = base + (n0 + n) * ld + m0 + mb;
  v8h o;
#pragma unroll
  for (int r = 0; r < 8; ++r) o[r] = (h16)f((int)(n0 + n), m0 + mb + r, acc[r]);
  *(v8h*)dst = o;
}
__device__ __forceinline__ void store_vecf(float* base, long ld, long n0, int m0, int lane,
                                           v8f acc) {
  int n = lane & 15, mb = (lane < 16) ? 0 : 8;
  *(v8f*)(base + (n0 + n) * ld + m0 + mb) = acc;
}

// ---- Tensor Data Mover: 1-D contiguous f16 run, global -> LDS ----
#if USE_TDM
__device__ __forceinline__ void tdm_load_1d_f16(h16* lds_dst, const h16* src, unsigned n) {
  unsigned long long ga = (unsigned long long)(uintptr_t)src;
  unsigned lds = (unsigned)(uintptr_t)lds_dst;  // generic LDS addr: low 32 bits = LDS offset
  u32x4 g0;
  g0[0] = 1u;                                            // count=1 (valid user D#)
  g0[1] = lds;                                           // lds_addr
  g0[2] = (unsigned)(ga & 0xFFFFFFFFu);                  // global_addr[31:0]
  g0[3] = (unsigned)((ga >> 32) & 0x1FFFFFFu) | (2u << 30);  // global_addr[56:32], type=2
  i32x8 g1;
  g1[0] = (int)(1u << 16);                               // mask=0, data_size=2B
  g1[1] = (int)((n & 0xFFFFu) << 16);                    // tensor_dim0[15:0] (bits 63:48)
  g1[2] = (int)(((n >> 16) & 0xFFFFu) | (1u << 16));     // tensor_dim0[31:16], tensor_dim1=1
  g1[3] = (int)(n << 16);                                // tile_dim0 (bits 127:112)
  g1[4] = 1;                                             // tile_dim1=1, tile_dim2=0
  g1[5] = (int)n;                                        // tensor_dim0_stride[31:0]
  g1[6] = 0;
  g1[7] = 0;
  i32x4 z4 = {0, 0, 0, 0};
  i32x8 z8 = {0, 0, 0, 0, 0, 0, 0, 0};
  __builtin_amdgcn_tensor_load_to_lds(g0, g1, z4, z4, z8, 0);
}
__device__ __forceinline__ void tdm_wait0() {
#if __has_builtin(__builtin_amdgcn_s_wait_tensorcnt)
  __builtin_amdgcn_s_wait_tensorcnt(0);
#else
  asm volatile("s_wait_tensorcnt 0x0" ::: "memory");
#endif
}
#endif

// ---------------- small / prep kernels ----------------

__global__ void k_embed1(const float* ext, const float* day_emb, const float* hour_emb,
                         const float* e1_w, const float* e1_b, float* g1) {
  int b = blockIdx.x, j = threadIdx.x;
  const float* e = ext + b * 6;
  int di = (int)e[2], hi = (int)e[3];
  float emb[6] = {e[0], day_emb[di * 2 + 0], day_emb[di * 2 + 1],
                  hour_emb[hi * 3 + 0], hour_emb[hi * 3 + 1], hour_emb[hi * 3 + 2]};
  float s = e1_b[j];
#pragma unroll
  for (int k = 0; k < 6; ++k) s += emb[k] * e1_w[j * 6 + k];
  g1[b * 64 + j] = geluf(s);
}

__global__ void k_embed2(const float* g1, const float* e2_w, const float* e2_b, float* g) {
  long t = (long)blockIdx.x * 256 + threadIdx.x;
  if (t >= NPIX) return;
  int p = (int)(t % HW);
  int b = (int)(t / HW);
  float s = e2_b[p];
  const float* w = e2_w + (long)p * 64;
  const float* gg = g1 + b * 64;
  for (int k = 0; k < 64; ++k) s += gg[k] * w[k];
  g[t] = geluf(s);
}

__global__ void k_encode(const float* inp, const float* g, const float* enc_w,
                         const float* enc_b, h16* x) {
  long t = (long)blockIdx.x * 256 + threadIdx.x;
  if (t >= NPIX * 64) return;
  int c = (int)(t & 63);
  long p = t >> 6;
  const float* w = enc_w + c * 5;
  const float* ip = inp + p * 4;
  float s = enc_b[c] + ip[0] * w[0] + ip[1] * w[1] + ip[2] * w[2] + ip[3] * w[3] + g[p] * w[4];
  x[t] = (h16)s;
}

__global__ void k_f32_to_f16(const float* src, h16* dst, long n) {
  long t = (long)blockIdx.x * 256 + threadIdx.x;
  if (t < n) dst[t] = (h16)src[t];
}

__global__ void k_prep_f3(const float* f3_w, h16* f3h) {
  int t = blockIdx.x * 256 + threadIdx.x;
  if (t >= 16 * 64) return;
  int m = t >> 6, k = t & 63;
  f3h[t] = (h16)((m < 2) ? f3_w[m * 64 + k] : 0.0f);
}

// Wre/Wim layout: [((l*24+kx)*12+ky)][o][i]
__global__ void k_prep_modes(const float* w1r, const float* w1i, const float* w2r,
                             const float* w2i, h16* Wre, h16* Wim) {
  long t = (long)blockIdx.x * 256 + threadIdx.x;
  long total = (long)LYR * KX * KY * 64 * 64;
  if (t >= total) return;
  long r = t;
  int i = (int)(r & 63); r >>= 6;
  int o = (int)(r & 63); r >>= 6;
  int ky = (int)(r % KY); r /= KY;
  int kx = (int)(r % KX);
  int l  = (int)(r / KX);
  int xi = (kx < 12) ? kx : kx - 12;
  long src = ((((long)l * 64 + i) * 64 + o) * 12 + xi) * 12 + ky;  // w[l][i][o][x][y]
  Wre[t] = (h16)((kx < 12) ? w1r[src] : w2r[src]);
  Wim[t] = (h16)((kx < 12) ? w1i[src] : w2i[src]);
}

// DFT / inverse-DFT matrices (f16): Fw 32x256, Fh 48x128, Gh 256x64, Gw 256x32
__global__ void k_gen_dft(h16* Fw, h16* Fh, h16* Gh, h16* Gw) {
  int t = blockIdx.x * 256 + threadIdx.x;
  if (t < 32 * 256) {
    int w = t & 255, m = t >> 8;
    float v = 0.f;
    if (m < 12) v = cosf(PI2 * m * w / 256.f);
    else if (m < 24) v = -sinf(PI2 * (m - 12) * w / 256.f);
    Fw[t] = (h16)v; return;
  }
  t -= 32 * 256;
  if (t < 48 * 128) {
    int h = t & 127, m = t >> 7;
    int j = (m < 24) ? m : m - 24;
    int kxs = (j < 12) ? j : j - 24;  // signed frequency
    float th = PI2 * kxs * h / 128.f;
    Fh[t] = (h16)((m < 24) ? cosf(th) : sinf(th)); return;
  }
  t -= 48 * 128;
  if (t < 256 * 64) {
    int k = t & 63, m = t >> 6;
    int h = (m < 128) ? m : m - 128;
    float v = 0.f;
    if (k < 48) {
      int j = (k < 24) ? k : k - 24;
      int kxs = (j < 12) ? j : j - 24;
      float th = PI2 * kxs * h / 128.f;
      float c = cosf(th) / 128.f, s = sinf(th) / 128.f;  // 1/H folded
      v = (m < 128) ? ((k < 24) ? c : -s) : ((k < 24) ? s : c);
    }
    Gh[t] = (h16)v; return;
  }
  t -= 256 * 64;
  if (t < 256 * 32) {
    int k = t & 31, w = t >> 5;
    float v = 0.f;
    if (k < 24) {
      int ky = (k < 12) ? k : k - 12;
      float cky = (ky == 0) ? 1.f : 2.f;   // rfft symmetry doubling
      float th = PI2 * ky * w / 256.f;
      v = (k < 12) ? cky * cosf(th) / 256.f : -cky * sinf(th) / 256.f;  // 1/W folded
    }
    Gw[t] = (h16)v; return;
  }
}

// Aconv[t][o][(dy*3+dx)*64+c]
__global__ void k_prep_conv(const float* tw, h16* Ac) {
  long t = (long)blockIdx.x * 256 + threadIdx.x;
  long total = (long)14 * 64 * 576;
  if (t >= total) return;
  int r = (int)(t % 576);
  long q = t / 576;
  int o = (int)(q & 63);
  int ti = (int)(q >> 6);
  int dd = r >> 6, c = r & 63;
  int dy = dd / 3, dx = dd - dy * 3;
  Ac[t] = (h16)tw[((((long)ti * 64 + o) * 64 + c) * 9) + dy * 3 + dx];
}

// ---------------- spectral path GEMMs ----------------

// Stage 1: DFT over w.  n=(b,h,c) N=262144, M=32 (24 used), K=256, Xw ld=32
__global__ void k_dft_w(const h16* x, const h16* Fw, h16* Xw) {
  int lane = threadIdx.x;
  long n0 = (long)blockIdx.x * 16;
  v8f acc[2];
#pragma unroll
  for (int mt = 0; mt < 2; ++mt) acc[mt] = (v8f){};
  for (int k0 = 0; k0 < 256; k0 += 32) {
    v16h bm = load_fragB(lane, [&](int n, int kk) {
      long nn = n0 + n;
      int c = (int)(nn & 63);
      long bh = nn >> 6;  // b*H + h
      return x[(bh * 256 + (k0 + kk)) * 64 + c];
    });
#pragma unroll
    for (int mt = 0; mt < 2; ++mt) {
      v16h a = loadA(Fw, 256, mt * 16, k0, lane);
      acc[mt] = wmma_acc(a, bm, acc[mt]);
    }
  }
#pragma unroll
  for (int mt = 0; mt < 2; ++mt)
    store_vec(Xw, 32, n0, mt * 16, lane, acc[mt], [](int, int, float v) { return v; });
}

// Stage 2: DFT over h. n=((b*64+c)*12+ky)*2+comp N=49152, M=48, K=128, P2 ld=48
__global__ void k_dft_h(const h16* Xw, const h16* Fh, float* P2) {
  int lane = threadIdx.x;
  long n0 = (long)blockIdx.x * 16;
  v8f acc[3];
#pragma unroll
  for (int mt = 0; mt < 3; ++mt) acc[mt] = (v8f){};
  for (int k0 = 0; k0 < 128; k0 += 32) {
    v16h bm = load_fragB(lane, [&](int n, int kk) {
      int nn = (int)(n0 + n);
      int comp = nn & 1;
      int q = nn >> 1;
      int ky = q % 12; q /= 12;
      int c = q & 63;
      int b = q >> 6;
      int h = k0 + kk;
      return Xw[((long)(b * 128 + h) * 64 + c) * 32 + comp * 12 + ky];
    });
#pragma unroll
    for (int mt = 0; mt < 3; ++mt) {
      v16h a = loadA(Fh, 128, mt * 16, k0, lane);
      acc[mt] = wmma_acc(a, bm, acc[mt]);
    }
  }
#pragma unroll
  for (int mt = 0; mt < 3; ++mt) store_vecf(P2, 48, n0, mt * 16, lane, acc[mt]);
}

// combine cos/sin products into complex spectrum, layout [mode][b][c]
// also emits a negated imaginary copy (f16 WMMA has no A/B-negate modifier)
__global__ void k_combine_fwd(const float* P2, h16* XfRe, h16* XfIm, h16* XfImN) {
  long t = (long)blockIdx.x * 256 + threadIdx.x;
  long total = (long)Bn * 64 * KX * KY;
  if (t >= total) return;
  int ky = (int)(t % 12); long q = t / 12;
  int kx = (int)(q % 24); q /= 24;
  int c = (int)(q & 63);
  int b = (int)(q >> 6);
  long br = (((long)(b * 64 + c) * 12 + ky) * 2 + 0) * 48;
  long bi = br + 48;
  float cxr = P2[br + kx], sxr = P2[br + 24 + kx];
  float cxi = P2[bi + kx], sxi = P2[bi + 24 + kx];
  long dst = ((long)(kx * 12 + ky) * 32 + b) * 64 + c;
  float re = cxr + sxi;
  float im = cxi - sxr;
  XfRe[dst]  = (h16)re;
  XfIm[dst]  = (h16)im;
  XfImN[dst] = (h16)(-im);
}

// complex channel mix per mode: Y[b,o] = sum_i Xf[b,i] * W[i,o]; M=64,K=64,N=32
__global__ void k_modemix(const h16* WreL, const h16* WimL, const h16* XfRe,
                          const h16* XfIm, const h16* XfImN, h16* Y) {
  int lane = threadIdx.x;
  int mode = blockIdx.x;
  int kx = mode / 12, ky = mode % 12;
  long n0 = (long)blockIdx.y * 16;
  const h16* Ar = WreL + (long)mode * 4096;
  const h16* Ai = WimL + (long)mode * 4096;
  const h16* Br = XfRe  + (long)mode * 32 * 64;
  const h16* Bi = XfIm  + (long)mode * 32 * 64;
  const h16* Bj = XfImN + (long)mode * 32 * 64;
  v8f accR[4], accI[4];
#pragma unroll
  for (int mt = 0; mt < 4; ++mt) { accR[mt] = (v8f){}; accI[mt] = (v8f){}; }
  for (int k0 = 0; k0 < 64; k0 += 32) {
    v16h br = loadB(Br, 64, n0, k0, lane);
    v16h bi = loadB(Bi, 64, n0, k0, lane);
    v16h bj = loadB(Bj, 64, n0, k0, lane);
#pragma unroll
    for (int mt = 0; mt < 4; ++mt) {
      v16h ar = loadA(Ar, 64, mt * 16, k0, lane);
      v16h ai = loadA(Ai, 64, mt * 16, k0, lane);
      accR[mt] = wmma_acc(ar, br, accR[mt]);
      accR[mt] = wmma_acc(ai, bj, accR[mt]);  // + Ai*(-Bi)
      accI[mt] = wmma_acc(ar, bi, accI[mt]);
      accI[mt] = wmma_acc(ai, br, accI[mt]);
    }
  }
#pragma unroll
  for (int mt = 0; mt < 4; ++mt) {
    store_frag(lane, accR[mt], [&](int n, int m, float v) {
      Y[(((long)(((int)n0 + n) * 64 + (mt * 16 + m)) * 12) + ky) * 64 + kx] = (h16)v;
    });
    store_frag(lane, accI[mt], [&](int n, int m, float v) {
      Y[(((long)(((int)n0 + n) * 64 + (mt * 16 + m)) * 12) + ky) * 64 + 24 + kx] = (h16)v;
    });
  }
}

// inverse over kx: n=(b,o,ky) N=24576, M=256 (Zre h<128 | Zim), K=64 (48 used)
__global__ void k_idft_h(const h16* Y, const h16* Gh, h16* P3) {
  int lane = threadIdx.x;
  long n0 = (long)blockIdx.x * 16;
  v16h b0 = loadB(Y, 64, n0, 0, lane);
  v16h b1 = loadB(Y, 64, n0, 32, lane);
#pragma unroll
  for (int mt = 0; mt < 16; ++mt) {
    v8f acc = {};
    acc = wmma_acc(loadA(Gh, 64, mt * 16, 0, lane), b0, acc);
    acc = wmma_acc(loadA(Gh, 64, mt * 16, 32, lane), b1, acc);
    store_vec(P3, 256, n0, mt * 16, lane, acc, [](int, int, float v) { return v; });
  }
}

// inverse over ky->w: n=(b,o,h) N=262144, M=256 (w), K=32 (24 used)
__global__ void k_idft_w(const h16* P3, const h16* Gw, h16* x1) {
  int lane = threadIdx.x;
  long n0 = (long)blockIdx.x * 16;
  v16h bm = load_fragB(lane, [&](int n, int kk) -> h16 {
    long nn = n0 + n;
    int h = (int)(nn & 127);
    long bo = nn >> 7;
    if (kk >= 24) return (h16)0.f;
    int ky = (kk < 12) ? kk : kk - 12;
    int comp = (kk < 12) ? 0 : 1;
    return P3[(bo * 12 + ky) * 256 + comp * 128 + h];
  });
#pragma unroll
  for (int mt = 0; mt < 16; ++mt) {
    v8f acc = {};
    acc = wmma_acc(loadA(Gw, 32, mt * 16, 0, lane), bm, acc);
    store_vec(x1, 256, n0, mt * 16, lane, acc, [](int, int, float v) { return v; });
  }
}

// skip GEMM fused with spectral add + bias + gelu. n=(b,h,w), M=64, K=64
__global__ void k_skip_combine(const h16* xin, const h16* Wsw, const float* wsb,
                               const h16* x1, h16* xout, int act) {
  int lane = threadIdx.x;
  long n0 = (long)blockIdx.x * 16;
  v16h b0 = loadB(xin, 64, n0, 0, lane);
  v16h b1 = loadB(xin, 64, n0, 32, lane);
#pragma unroll
  for (int mt = 0; mt < 4; ++mt) {
    v8f acc = {};
    acc = wmma_acc(loadA(Wsw, 64, mt * 16, 0, lane), b0, acc);
    acc = wmma_acc(loadA(Wsw, 64, mt * 16, 32, lane), b1, acc);
    store_vec(xout, 64, n0, mt * 16, lane, acc, [&](int nn, int o, float v) {
      int w = nn & 255, h = (nn >> 8) & 127, b = nn >> 15;
      float s = v + wsb[o] + (float)x1[((long)(b * 64 + o) * 128 + h) * 256 + w];
      return act ? geluf(s) : s;
    });
  }
}

// 3x3 conv, per-batch weight bank, implicit GEMM K=576
__global__ void k_conv(const h16* x, const h16* Aconv, const float* tb, const float* ext,
                       h16* hfeat) {
  int lane = threadIdx.x;
  long n0 = (long)blockIdx.x * 16;
  int b = (int)(n0 >> 15);
  int ti = (int)ext[b * 6 + 3] - 5;
  const h16* A = Aconv + (long)ti * 64 * 576;
  const float* bias = tb + ti * 64;
  v8f acc[4];
#pragma unroll
  for (int mt = 0; mt < 4; ++mt) acc[mt] = (v8f){};
  // per-lane pixel for B fragments
  int nlane = (int)(n0 & 32767) + (lane & 15);
  int w = nlane & 255, h = (nlane >> 8) & 127;
  // warm L2 for the next tile's pixel row while we grind through K=576
  __builtin_prefetch(x + ((n0 + 16) & (NPIX - 1)) * 64, 0, 1);
  for (int k0 = 0; k0 < 576; k0 += 32) {
    int k = k0 + ((lane < 16) ? 0 : 16);  // start of this lane's 16-half run
    int c0 = k & 63, dd = k >> 6;         // run never crosses a (dy,dx) boundary
    int dy = dd / 3, dx = dd - dy * 3;
    int hy = h + dy - 1, wx = w + dx - 1;
    v16h bm;
    if (hy >= 0 && hy < 128 && wx >= 0 && wx < 256) {
      bm = *(const v16h*)(x + ((long)(b * 128 + hy) * 256 + wx) * 64 + c0);
    } else {
      v16h z = {};
      bm = z;
    }
#pragma unroll
    for (int mt = 0; mt < 4; ++mt) {
      v16h a = loadA(A, 576, mt * 16, k0, lane);
      acc[mt] = wmma_acc(a, bm, acc[mt]);
    }
  }
#pragma unroll
  for (int mt = 0; mt < 4; ++mt) {
    store_vec(hfeat, 64, n0, mt * 16, lane, acc[mt],
              [&](int, int o, float v) { return geluf(v + bias[o]); });
  }
}

// fused per-pixel MLP 64->128->64->2, z tiles staged in LDS
#define NWAVE 4
__global__ void __launch_bounds__(128) k_mlp(const h16* hfeat, const h16* f1h,
                                             const float* f1b, const h16* f2h,
                                             const float* f2b, const h16* f3h,
                                             const float* f3b, float* out) {
  __shared__ h16 sW1[128 * 64];
  __shared__ h16 sW2[64 * 128];
  __shared__ h16 sW3[16 * 64];
  __shared__ h16 sz1[NWAVE][16 * 128];
  __shared__ h16 sz2[NWAVE][16 * 64];
  int tid = threadIdx.x;
  int wave = tid >> 5, lane = tid & 31;
#if USE_TDM
  if (wave == 0) {
    tdm_load_1d_f16(sW1, f1h, 128 * 64);
    tdm_load_1d_f16(sW2, f2h, 64 * 128);
    tdm_load_1d_f16(sW3, f3h, 16 * 64);
    tdm_wait0();
  }
#else
  for (int i = tid; i < 128 * 64; i += 128) sW1[i] = f1h[i];
  for (int i = tid; i < 64 * 128; i += 128) sW2[i] = f2h[i];
  for (int i = tid; i < 16 * 64; i += 128) sW3[i] = f3h[i];
#endif
  __syncthreads();
  long n0 = ((long)blockIdx.x * NWAVE + wave) * 16;
  h16* z1 = sz1[wave];
  h16* z2 = sz2[wave];
  {
    v16h b0 = loadB(hfeat, 64, n0, 0, lane);
    v16h b1 = loadB(hfeat, 64, n0, 32, lane);
#pragma unroll
    for (int mt = 0; mt < 8; ++mt) {
      v8f acc = {};
      acc = wmma_acc(loadA(sW1, 64, mt * 16, 0, lane), b0, acc);
      acc = wmma_acc(loadA(sW1, 64, mt * 16, 32, lane), b1, acc);
      store_vec(z1, 128, 0, mt * 16, lane, acc,
                [&](int, int o, float v) { return geluf(v + f1b[o]); });
    }
  }
  asm volatile("s_wait_dscnt 0x0" ::: "memory");
  {
    v16h b0 = loadB(z1, 128, 0, 0, lane);
    v16h b1 = loadB(z1, 128, 0, 32, lane);
    v16h b2 = loadB(z1, 128, 0, 64, lane);
    v16h b3 = loadB(z1, 128, 0, 96, lane);
#pragma unroll
    for (int mt = 0; mt < 4; ++mt) {
      v8f acc = {};
      acc = wmma_acc(loadA(sW2, 128, mt * 16, 0, lane), b0, acc);
      acc = wmma_acc(loadA(sW2, 128, mt * 16, 32, lane), b1, acc);
      acc = wmma_acc(loadA(sW2, 128, mt * 16, 64, lane), b2, acc);
      acc = wmma_acc(loadA(sW2, 128, mt * 16, 96, lane), b3, acc);
      store_vec(z2, 64, 0, mt * 16, lane, acc,
                [&](int, int o, float v) { return geluf(v + f2b[o]); });
    }
  }
  asm volatile("s_wait_dscnt 0x0" ::: "memory");
  {
    v16h b0 = loadB(z2, 64, 0, 0, lane);
    v16h b1 = loadB(z2, 64, 0, 32, lane);
    v8f acc = {};
    acc = wmma_acc(loadA(sW3, 64, 0, 0, lane), b0, acc);
    acc = wmma_acc(loadA(sW3, 64, 0, 32, lane), b1, acc);
    store_frag(lane, acc, [&](int n, int m, float v) {
      if (m < 2) out[(n0 + n) * 2 + m] = v + f3b[m];
    });
  }
}

// ---------------- launcher ----------------

extern "C" void kernel_launch(void* const* d_in, const int* in_sizes, int n_in,
                              void* d_out, int out_size, void* d_ws, size_t ws_size,
                              hipStream_t stream) {
  (void)in_sizes; (void)n_in; (void)out_size; (void)ws_size;
  const float* inp   = (const float*)d_in[0];
  const float* ext   = (const float*)d_in[1];
  const float* enc_w = (const float*)d_in[2];
  const float* enc_b = (const float*)d_in[3];
  const float* w1r   = (const float*)d_in[4];
  const float* w1i   = (const float*)d_in[5];
  const float* w2r   = (const float*)d_in[6];
  const float* w2i   = (const float*)d_in[7];
  const float* ws_w  = (const float*)d_in[8];
  const float* ws_b  = (const float*)d_in[9];
  const float* day_emb  = (const float*)d_in[10];
  const float* hour_emb = (const float*)d_in[11];
  const float* e1_w = (const float*)d_in[12];
  const float* e1_b = (const float*)d_in[13];
  const float* e2_w = (const float*)d_in[14];
  const float* e2_b = (const float*)d_in[15];
  const float* tw   = (const float*)d_in[16];
  const float* tb   = (const float*)d_in[17];
  const float* f1_w = (const float*)d_in[18];
  const float* f1_b = (const float*)d_in[19];
  const float* f2_w = (const float*)d_in[20];
  const float* f2_b = (const float*)d_in[21];
  const float* f3_w = (const float*)d_in[22];
  const float* f3_b = (const float*)d_in[23];
  float* out = (float*)d_out;

  char* p = (char*)d_ws;
  auto alloc = [&](size_t bytes) -> char* {
    char* r = p;
    p += (bytes + 255) & ~(size_t)255;
    return r;
  };
  float* g1  = (float*)alloc(32 * 64 * 4);
  float* g   = (float*)alloc((size_t)NPIX * 4);
  h16* xA    = (h16*)alloc((size_t)NPIX * 64 * 2);
  h16* xB    = (h16*)alloc((size_t)NPIX * 64 * 2);
  h16* x1    = (h16*)alloc((size_t)NPIX * 64 * 2);  // also reused as hfeat
  h16* Xw    = (h16*)alloc((size_t)Bn * Hh * 64 * 32 * 2);  // ld=32 (24 used)
  float* P2  = (float*)alloc((size_t)Bn * 64 * 12 * 2 * 48 * 4);
  h16* XfRe  = (h16*)alloc((size_t)288 * 32 * 64 * 2);
  h16* XfIm  = (h16*)alloc((size_t)288 * 32 * 64 * 2);
  h16* XfImN = (h16*)alloc((size_t)288 * 32 * 64 * 2);
  h16* Ybuf  = (h16*)alloc((size_t)Bn * 64 * 12 * 64 * 2);
  h16* P3    = (h16*)alloc((size_t)Bn * 64 * 12 * 256 * 2);
  h16* wswh  = (h16*)alloc((size_t)LYR * 64 * 64 * 2);
  h16* Wre   = (h16*)alloc((size_t)LYR * KX * KY * 64 * 64 * 2);
  h16* Wim   = (h16*)alloc((size_t)LYR * KX * KY * 64 * 64 * 2);
  h16* Fw    = (h16*)alloc(32 * 256 * 2);
  h16* Fh    = (h16*)alloc(48 * 128 * 2);
  h16* Gh    = (h16*)alloc(256 * 64 * 2);
  h16* Gw    = (h16*)alloc(256 * 32 * 2);
  h16* Aconv = (h16*)alloc((size_t)14 * 64 * 576 * 2);
  h16* f1h   = (h16*)alloc(128 * 64 * 2);
  h16* f2h   = (h16*)alloc(64 * 128 * 2);
  h16* f3h   = (h16*)alloc(16 * 64 * 2);

  (void)hipMemsetAsync(Ybuf, 0, (size_t)Bn * 64 * 12 * 64 * 2, stream);

  k_embed1<<<Bn, 64, 0, stream>>>(ext, day_emb, hour_emb, e1_w, e1_b, g1);
  k_embed2<<<(int)((NPIX + 255) / 256), 256, 0, stream>>>(g1, e2_w, e2_b, g);
  k_encode<<<(int)((NPIX * 64 + 255) / 256), 256, 0, stream>>>(inp, g, enc_w, enc_b, xA);
  k_f32_to_f16<<<(LYR * 4096 + 255) / 256, 256, 0, stream>>>(ws_w, wswh, LYR * 4096);
  k_f32_to_f16<<<(128 * 64 + 255) / 256, 256, 0, stream>>>(f1_w, f1h, 128 * 64);
  k_f32_to_f16<<<(64 * 128 + 255) / 256, 256, 0, stream>>>(f2_w, f2h, 64 * 128);
  k_prep_f3<<<4, 256, 0, stream>>>(f3_w, f3h);
  {
    long tot = (long)LYR * KX * KY * 64 * 64;
    k_prep_modes<<<(int)((tot + 255) / 256), 256, 0, stream>>>(w1r, w1i, w2r, w2i, Wre, Wim);
  }
  k_gen_dft<<<(38912 + 255) / 256, 256, 0, stream>>>(Fw, Fh, Gh, Gw);
  {
    long tot = (long)14 * 64 * 576;
    k_prep_conv<<<(int)((tot + 255) / 256), 256, 0, stream>>>(tw, Aconv);
  }

  for (int l = 0; l < LYR; ++l) {
    h16* xin  = (l & 1) ? xB : xA;
    h16* xout = (l & 1) ? xA : xB;
    k_dft_w<<<16384, 32, 0, stream>>>(xin, Fw, Xw);
    k_dft_h<<<3072, 32, 0, stream>>>(Xw, Fh, P2);
    {
      long tot = (long)Bn * 64 * 288;
      k_combine_fwd<<<(int)((tot + 255) / 256), 256, 0, stream>>>(P2, XfRe, XfIm, XfImN);
    }
    k_modemix<<<dim3(288, 2), 32, 0, stream>>>(Wre + (long)l * 288 * 4096,
                                               Wim + (long)l * 288 * 4096, XfRe, XfIm, XfImN,
                                               Ybuf);
    k_idft_h<<<1536, 32, 0, stream>>>(Ybuf, Gh, P3);
    k_idft_w<<<16384, 32, 0, stream>>>(P3, Gw, x1);
    k_skip_combine<<<65536, 32, 0, stream>>>(xin, wswh + l * 4096, ws_b + l * 64, x1, xout,
                                             (l < LYR - 1) ? 1 : 0);
  }
  // after 4 layers the activation lives in xA
  h16* hfeat = x1;
  k_conv<<<65536, 32, 0, stream>>>(xA, Aconv, tb, ext, hfeat);
  k_mlp<<<(int)(NPIX / (16 * NWAVE)), 128, 0, stream>>>(hfeat, f1h, f1_b, f2h, f2_b, f3h, f3_b,
                                                        out);
}
